// MoE_20289425506608
// MI455X (gfx1250) — matrix operand order repaired and checked by reading, not compiled, not run on hardware
//
#include <hip/hip_runtime.h>
#include <cstdint>

// ---- problem constants (from reference: B=4, T=2048, C=1024, E=8, K=2, NS=1) ----
#define B_   4
#define T_   2048
#define C_   1024
#define E_   8
#define K_   2
#define FF_  4096
#define BT_  8192
#define CAP_ 1024

typedef __bf16 bf16;
typedef __attribute__((ext_vector_type(16))) bf16  bf16x16;
typedef __attribute__((ext_vector_type(8)))  float floatx8;
typedef __attribute__((ext_vector_type(4)))  unsigned int uint32x4;
typedef __attribute__((ext_vector_type(8)))  int int32x8;
typedef __attribute__((ext_vector_type(4)))  int int32x4;

union FragAB { bf16x16 v; uint4 q[2]; };
union Pack4  { bf16 h[4]; uint2 u; };

#if defined(__has_builtin)
#if __has_builtin(__builtin_amdgcn_tensor_load_to_lds) && __has_builtin(__builtin_amdgcn_s_wait_tensorcnt)
#define HAVE_TDM 1
#endif
#endif
#ifndef HAVE_TDM
#define HAVE_TDM 0
#endif

// ---------------------------------------------------------------------------
// init: slot_token = -1, p_accum = 0
// ---------------------------------------------------------------------------
__global__ void init_kernel(int* __restrict__ slot_token, float* __restrict__ p_accum) {
    int i = blockIdx.x * 256 + threadIdx.x;
    if (i < E_ * CAP_) slot_token[i] = -1;
    if (i < E_) p_accum[i] = 0.0f;
}

// ---------------------------------------------------------------------------
// transpose + f32 -> bf16 convert:  src [e][Kd][Nd]  ->  dst [e][Nd][Kd]
// ---------------------------------------------------------------------------
__global__ __launch_bounds__(256) void transpose_convert_kernel(
    const float* __restrict__ src, bf16* __restrict__ dst, int Kd, int Nd) {
    __shared__ float tile[64][65];
    const float* s = src + (size_t)blockIdx.z * Kd * Nd;
    bf16*        d = dst + (size_t)blockIdx.z * Kd * Nd;
    int n0 = blockIdx.x * 64, k0 = blockIdx.y * 64;
    int col = threadIdx.x & 63, rbase = threadIdx.x >> 6;
#pragma unroll
    for (int rr = 0; rr < 16; ++rr) {
        int r = rbase + rr * 4;
        tile[r][col] = s[(size_t)(k0 + r) * Nd + (n0 + col)];
    }
    __syncthreads();
#pragma unroll
    for (int rr = 0; rr < 16; ++rr) {
        int r = rbase + rr * 4;
        d[(size_t)(n0 + r) * Kd + (k0 + col)] = (bf16)tile[col][r];
    }
}

// ---------------------------------------------------------------------------
// router: one wave32 per token
// ---------------------------------------------------------------------------
__global__ __launch_bounds__(256) void router_kernel(
    const float* __restrict__ x, const float* __restrict__ Wc, const float* __restrict__ bc,
    float* __restrict__ gates, int* __restrict__ topk, float* __restrict__ sc0,
    float* __restrict__ p_accum) {
    int wave = threadIdx.x >> 5, lane = threadIdx.x & 31;
    int t = blockIdx.x * 8 + wave;
    const float* xr = x + (size_t)t * C_;
    float acc[E_];
#pragma unroll
    for (int e = 0; e < E_; ++e) acc[e] = 0.0f;
    for (int c = lane; c < C_; c += 32) {
        float xv = xr[c];
        const float* wr = Wc + (size_t)c * E_;
#pragma unroll
        for (int e = 0; e < E_; ++e) acc[e] += xv * wr[e];
    }
#pragma unroll
    for (int e = 0; e < E_; ++e) {
#pragma unroll
        for (int off = 16; off > 0; off >>= 1) acc[e] += __shfl_xor(acc[e], off, 32);
    }
    if (lane == 0) {
        float s[E_]; float nsum = 0.0f;
#pragma unroll
        for (int e = 0; e < E_; ++e) { s[e] = 1.0f / (1.0f + expf(-(acc[e] + bc[e]))); nsum += s[e]; }
        int i0 = 0;
#pragma unroll
        for (int e = 1; e < E_; ++e) if (s[e] > s[i0]) i0 = e;
        int i1 = (i0 == 0) ? 1 : 0;
#pragma unroll
        for (int e = 0; e < E_; ++e) if (e != i0 && s[e] > s[i1]) i1 = e;
        float g0 = s[i0], g1 = s[i1], inv = 1.0f / (g0 + g1);
        gates[t * K_ + 0] = g0 * inv;
        gates[t * K_ + 1] = g1 * inv;
        topk[t * K_ + 0] = i0;
        topk[t * K_ + 1] = i1;
        sc0[t] = g0 * inv;
        atomicAdd(&p_accum[i0], s[i0] / nsum);
        atomicAdd(&p_accum[i1], s[i1] / nsum);
    }
}

// ---------------------------------------------------------------------------
// stable descending bitonic argsort of 8192 gate values
// ---------------------------------------------------------------------------
__global__ __launch_bounds__(1024) void sort_kernel(
    const float* __restrict__ sc0, int* __restrict__ order) {
    __shared__ float key[BT_];
    __shared__ int   idx[BT_];
    for (int i = threadIdx.x; i < BT_; i += 1024) { key[i] = sc0[i]; idx[i] = i; }
    __syncthreads();
    for (int k = 2; k <= BT_; k <<= 1) {
        for (int j = k >> 1; j > 0; j >>= 1) {
            for (int i = threadIdx.x; i < BT_; i += 1024) {
                int ixj = i ^ j;
                if (ixj > i) {
                    bool desc = ((i & k) == 0);
                    float ka = key[i], kb = key[ixj];
                    int   ia = idx[i], ib = idx[ixj];
                    bool aFirst = (ka > kb) || (ka == kb && ia < ib);
                    if (desc ? !aFirst : aFirst) {
                        key[i] = kb; key[ixj] = ka; idx[i] = ib; idx[ixj] = ia;
                    }
                }
            }
            __syncthreads();
        }
    }
    for (int i = threadIdx.x; i < BT_; i += 1024) order[i] = idx[i];
}

// ---------------------------------------------------------------------------
// capacity-aware slot assignment (histogram + block prefix scan + replay)
// ---------------------------------------------------------------------------
__global__ __launch_bounds__(1024) void assign_kernel(
    const int* __restrict__ order, const int* __restrict__ topk,
    int* __restrict__ token_slot, int* __restrict__ slot_token, int* __restrict__ counts) {
    __shared__ unsigned bufA[1024 * E_];
    __shared__ unsigned bufB[1024 * E_];
    int tid = threadIdx.x;
    unsigned cnt[E_];
#pragma unroll
    for (int e = 0; e < E_; ++e) cnt[e] = 0;
    int e_of[16];
#pragma unroll
    for (int q = 0; q < 16; ++q) {
        int j = tid * 16 + q;
        int k = j >> 13;
        int i = j & (BT_ - 1);
        int t = order[i];
        int e = topk[t * K_ + k];
        e_of[q] = e;
        cnt[e]++;
    }
#pragma unroll
    for (int e = 0; e < E_; ++e) bufA[tid * E_ + e] = cnt[e];
    __syncthreads();
    unsigned* src = bufA; unsigned* dst = bufB;
    for (int off = 1; off < 1024; off <<= 1) {
#pragma unroll
        for (int e = 0; e < E_; ++e) {
            unsigned v = src[tid * E_ + e];
            if (tid >= off) v += src[(tid - off) * E_ + e];
            dst[tid * E_ + e] = v;
        }
        __syncthreads();
        unsigned* tmp = src; src = dst; dst = tmp;
    }
    unsigned base[E_];
#pragma unroll
    for (int e = 0; e < E_; ++e) base[e] = (tid > 0) ? src[(tid - 1) * E_ + e] : 0u;
    if (tid == 1023) {
#pragma unroll
        for (int e = 0; e < E_; ++e) counts[e] = (int)src[1023 * E_ + e];
    }
#pragma unroll
    for (int q = 0; q < 16; ++q) {
        int j = tid * 16 + q;
        int k = j >> 13;
        int i = j & (BT_ - 1);
        int t = order[i];
        int e = e_of[q];
        unsigned slot = base[e]++;
        if (slot < (unsigned)CAP_) {
            token_slot[t * K_ + k] = e * CAP_ + (int)slot;
            slot_token[e * CAP_ + (int)slot] = t * K_ + k;
        } else {
            token_slot[t * K_ + k] = -1;
        }
    }
}

// ---------------------------------------------------------------------------
// gather dispatched tokens into bf16 expert_in [E*CAP, C]
// ---------------------------------------------------------------------------
__global__ __launch_bounds__(256) void gather_kernel(
    const float* __restrict__ x, const int* __restrict__ slot_token,
    bf16* __restrict__ expert_in) {
    int s = blockIdx.x;
    int tk = slot_token[s];
    int c = threadIdx.x * 4;
    Pack4 p;
    if (tk >= 0) {
        int t = tk >> 1;
        float4 v = *(const float4*)(x + (size_t)t * C_ + c);
        p.h[0] = (bf16)v.x; p.h[1] = (bf16)v.y; p.h[2] = (bf16)v.z; p.h[3] = (bf16)v.w;
    } else {
        p.u.x = 0u; p.u.y = 0u;
    }
    *(uint2*)(expert_in + (size_t)s * C_ + c) = p.u;
}

// ---------------------------------------------------------------------------
// TDM tile load: 2D tile (rows x 32 bf16 elems), row stride = strideElems,
// DMA'd into LDS with 4-DWORD padding every 16 DWORDs (=> 40-half LDS rows,
// matching the bank-conflict-free fragment layout). D# packing per
// CDNA5 ISA (group0/group1; groups 2,3 unused for 2D).
// This toolchain's builtin takes 6 args (extra int32x8 group slot + cpol).
// ---------------------------------------------------------------------------
#define TK_  32
#define LDA_ 40   // padded halves per LDS row

#if HAVE_TDM
__device__ __forceinline__ void tdm_load_tile(const bf16* gptr, unsigned lds_off,
                                              unsigned rows, unsigned long long strideElems) {
    unsigned long long ga = (unsigned long long)(uintptr_t)gptr;
    uint32x4 g0;
    g0[0] = 1u;                                               // count=1 (user descriptor)
    g0[1] = lds_off;                                          // lds_addr (bytes)
    g0[2] = (unsigned)(ga & 0xffffffffu);                     // global_addr[31:0]
    g0[3] = (unsigned)((ga >> 32) & 0x01ffffffu) | (2u << 30);// global_addr[56:32], type=2
    const unsigned td0 = TK_;                                 // tensor_dim0 (elems)
    const unsigned td1 = rows;                                // tensor_dim1
    int32x8 g1;
    g1[0] = (int)((1u << 16)   |                              // data_size = 2 bytes
                  (1u << 20)   |                              // pad_enable
                  (3u << 22)   |                              // pad_interval: 16 DWORDs
                  (3u << 25));                                // pad_amount:   4 DWORDs
    g1[1] = (int)((td0 & 0xffffu) << 16);                     // tensor_dim0[15:0]
    g1[2] = (int)(((td0 >> 16) & 0xffffu) | ((td1 & 0xffffu) << 16));
    g1[3] = (int)(((td1 >> 16) & 0xffffu) | ((unsigned)TK_ << 16)); // tile_dim0 = 32
    g1[4] = (int)(rows & 0xffffu);                            // tile_dim1 = rows, tile_dim2 = 0
    g1[5] = (int)(unsigned)(strideElems & 0xffffffffull);     // tensor_dim0_stride[31:0]
    g1[6] = (int)(unsigned)((strideElems >> 32) & 0xffffull); // tensor_dim0_stride[47:32]
    g1[7] = 0;
    int32x4 gz4; gz4[0] = 0; gz4[1] = 0; gz4[2] = 0; gz4[3] = 0;  // groups 2,3 unused (2D)
    int32x8 gz8;
#pragma unroll
    for (int i = 0; i < 8; ++i) gz8[i] = 0;                   // extra group slot (clang-23 form)
    __builtin_amdgcn_tensor_load_to_lds(g0, g1, gz4, gz4, gz8, 0);
}
#endif

// ---------------------------------------------------------------------------
// bf16 WMMA GEMM: out[e] = A[e] (MxK, K-contig) * B[e]^T (NxK, K-contig) + bias
// MODE 0: +b1 -> tanh-GELU -> bf16 out ; MODE 1: +b2 -> f32 out
// 256 thr = 8 waves (2Mx4N), block tile 128x128, K-step 32.
// Staging: TDM double-buffered (wave 0 issues, s_wait_tensorcnt) when
// available; manual global->LDS copies otherwise.
// ---------------------------------------------------------------------------
__device__ __forceinline__ float gelu_tanh(float v) {
    float v3 = v * v * v;
    return 0.5f * v * (1.0f + tanhf(0.7978845608028654f * (v + 0.044715f * v3)));
}

template <int MODE>
__global__ __launch_bounds__(256) void gemm_bf16_kernel(
    const bf16* __restrict__ Ab, const bf16* __restrict__ Bb,
    const float* __restrict__ biasAll, void* __restrict__ outp,
    int N, int Kd, long aStride, long bStride, long oStride, int biasStride) {
    __shared__ __attribute__((aligned(16))) bf16 sA[2][128 * LDA_];
    __shared__ __attribute__((aligned(16))) bf16 sB[2][128 * LDA_];
    int e = blockIdx.z;
    const bf16* A = Ab + (size_t)aStride * e;
    const bf16* B = Bb + (size_t)bStride * e;
    const float* bias = biasAll + (size_t)biasStride * e;
    int n0 = blockIdx.x * 128, m0 = blockIdx.y * 128;
    int tid = threadIdx.x;
    int lane = tid & 31, w = tid >> 5;
    int wm = w >> 2, wn = w & 3;
    int hh = lane >> 4, ml = lane & 15;

    floatx8 acc[4][2];
#pragma unroll
    for (int s = 0; s < 4; ++s)
#pragma unroll
        for (int u = 0; u < 2; ++u)
#pragma unroll
            for (int i = 0; i < 8; ++i) acc[s][u][i] = 0.0f;

    const int nK = Kd / TK_;

#if HAVE_TDM
    unsigned ldsA[2] = { (unsigned)(uintptr_t)&sA[0][0], (unsigned)(uintptr_t)&sA[1][0] };
    unsigned ldsB[2] = { (unsigned)(uintptr_t)&sB[0][0], (unsigned)(uintptr_t)&sB[1][0] };
    if (tid < 32) {   // wave 0 drives the Tensor Data Mover
        tdm_load_tile(A + (size_t)m0 * Kd, ldsA[0], 128, (unsigned long long)Kd);
        tdm_load_tile(B + (size_t)n0 * Kd, ldsB[0], 128, (unsigned long long)Kd);
    }
#else
    int ldRow = tid >> 1;
    int ldHalf = (tid & 1) * 16;
#endif

    for (int ki = 0; ki < nK; ++ki) {
        int buf = ki & 1;
        int kk = ki * TK_;
#if HAVE_TDM
        if (tid < 32) {
            if (ki + 1 < nK) {   // prefetch next tile into the other buffer
                int kn = kk + TK_;
                tdm_load_tile(A + (size_t)m0 * Kd + kn, ldsA[buf ^ 1], 128, (unsigned long long)Kd);
                tdm_load_tile(B + (size_t)n0 * Kd + kn, ldsB[buf ^ 1], 128, (unsigned long long)Kd);
                __builtin_amdgcn_s_wait_tensorcnt(2);  // in-order: current buffer landed
            } else {
                __builtin_amdgcn_s_wait_tensorcnt(0);
            }
        }
        __syncthreads();
#else
        const bf16* gA = &A[(size_t)(m0 + ldRow) * Kd + kk + ldHalf];
        const bf16* gB = &B[(size_t)(n0 + ldRow) * Kd + kk + ldHalf];
        *(uint4*)&sA[buf][ldRow * LDA_ + ldHalf]     = *(const uint4*)gA;
        *(uint4*)&sA[buf][ldRow * LDA_ + ldHalf + 8] = *(const uint4*)(gA + 8);
        *(uint4*)&sB[buf][ldRow * LDA_ + ldHalf]     = *(const uint4*)gB;
        *(uint4*)&sB[buf][ldRow * LDA_ + ldHalf + 8] = *(const uint4*)(gB + 8);
        __syncthreads();
#endif
        FragAB bfr[2];
#pragma unroll
        for (int u = 0; u < 2; ++u) {
            int nrow = wn * 32 + u * 16 + ml;      // B: lane n = lane%16; K = 16*hh..+15
            bfr[u].q[0] = *(const uint4*)&sB[buf][nrow * LDA_ + hh * 16];
            bfr[u].q[1] = *(const uint4*)&sB[buf][nrow * LDA_ + hh * 16 + 8];
        }
#pragma unroll
        for (int s = 0; s < 4; ++s) {
            FragAB afr;
            int mrow = wm * 64 + s * 16 + ml;      // A: K = 16*hh..+15
            afr.q[0] = *(const uint4*)&sA[buf][mrow * LDA_ + hh * 16];
            afr.q[1] = *(const uint4*)&sA[buf][mrow * LDA_ + hh * 16 + 8];
#pragma unroll
            for (int u = 0; u < 2; ++u)
                acc[s][u] = __builtin_amdgcn_wmma_f32_16x16x32_bf16(
                    false, afr.v, false, bfr[u].v, (short)0, acc[s][u], false, false);
        }
        __syncthreads();
    }

#pragma unroll
    for (int u = 0; u < 2; ++u) {
        int n = n0 + wn * 32 + u * 16 + ml;
        float bv = bias[n];
#pragma unroll
        for (int s = 0; s < 4; ++s) {
            int mBase = m0 + wm * 64 + s * 16 + hh * 8;
#pragma unroll
            for (int v = 0; v < 8; ++v) {
                float val = acc[s][u][v] + bv;
                size_t idx = (size_t)oStride * e + (size_t)(mBase + v) * N + n;
                if (MODE == 0) {
                    ((bf16*)outp)[idx] = (bf16)gelu_tanh(val);
                } else {
                    ((float*)outp)[idx] = val;
                }
            }
        }
    }
}

// ---------------------------------------------------------------------------
// combine: out[t] = sum_k gate_k * expert_out[slot_k]   (dropped -> 0)
// ---------------------------------------------------------------------------
__global__ __launch_bounds__(256) void combine_kernel(
    const float* __restrict__ eout, const int* __restrict__ token_slot,
    const float* __restrict__ gates, float* __restrict__ out) {
    int t = blockIdx.x;
    int c = threadIdx.x * 4;
    float4 r; r.x = 0.f; r.y = 0.f; r.z = 0.f; r.w = 0.f;
    int s0 = token_slot[t * K_ + 0], s1 = token_slot[t * K_ + 1];
    float g0 = gates[t * K_ + 0], g1 = gates[t * K_ + 1];
    if (s0 >= 0) {
        float4 v = *(const float4*)(eout + (size_t)s0 * C_ + c);
        r.x += g0 * v.x; r.y += g0 * v.y; r.z += g0 * v.z; r.w += g0 * v.w;
    }
    if (s1 >= 0) {
        float4 v = *(const float4*)(eout + (size_t)s1 * C_ + c);
        r.x += g1 * v.x; r.y += g1 * v.y; r.z += g1 * v.z; r.w += g1 * v.w;
    }
    *(float4*)(out + (size_t)t * C_ + c) = r;
}

// ---------------------------------------------------------------------------
// stats tail: [tokens_per_expert(8) | f(8) | p(8)]
// ---------------------------------------------------------------------------
__global__ void stats_kernel(const int* __restrict__ counts,
                             const float* __restrict__ p_accum,
                             float* __restrict__ tail) {
    int e = threadIdx.x;
    if (e < E_) {
        float fv = (float)counts[e] / (float)BT_;
        tail[e] = fv;
        tail[E_ + e] = fv;
        tail[2 * E_ + e] = p_accum[e] / (float)BT_;
    }
}

// ---------------------------------------------------------------------------
extern "C" void kernel_launch(void* const* d_in, const int* in_sizes, int n_in,
                              void* d_out, int out_size, void* d_ws, size_t ws_size,
                              hipStream_t stream) {
    (void)in_sizes; (void)n_in; (void)out_size; (void)ws_size;
    const float* x  = (const float*)d_in[0];
    // d_in[1]=Ws, d_in[2]=bs : shared-expert branch is dead in the reference output
    const float* Wc = (const float*)d_in[3];
    const float* bc = (const float*)d_in[4];
    const float* W1 = (const float*)d_in[5];
    const float* b1 = (const float*)d_in[6];
    const float* W2 = (const float*)d_in[7];
    const float* b2 = (const float*)d_in[8];
    // d_in[9] = capacity = B*T/E = 1024 (compile-time constant CAP_)

    size_t off = 0;
    char* ws = (char*)d_ws;
    auto take = [&](size_t bytes) -> void* {
        void* p = ws + off;
        off = (off + bytes + 255) & ~(size_t)255;
        return p;
    };
    float* gates      = (float*)take((size_t)BT_ * K_ * 4);
    int*   topk       = (int*)  take((size_t)BT_ * K_ * 4);
    float* sc0        = (float*)take((size_t)BT_ * 4);
    int*   order      = (int*)  take((size_t)BT_ * 4);
    int*   token_slot = (int*)  take((size_t)BT_ * K_ * 4);
    int*   slot_token = (int*)  take((size_t)E_ * CAP_ * 4);
    int*   counts     = (int*)  take(E_ * 4);
    float* p_accum    = (float*)take(E_ * 4);
    bf16*  expert_in  = (bf16*) take((size_t)E_ * CAP_ * C_ * 2);
    bf16*  hbuf       = (bf16*) take((size_t)E_ * CAP_ * FF_ * 2);
    float* eout       = (float*)take((size_t)E_ * CAP_ * C_ * 4);
    bf16*  W1T        = (bf16*) take((size_t)E_ * FF_ * C_ * 2);
    bf16*  W2T        = (bf16*) take((size_t)E_ * C_ * FF_ * 2);

    float* out  = (float*)d_out;
    float* tail = out + (size_t)BT_ * C_;

    init_kernel<<<(E_ * CAP_ + 255) / 256, 256, 0, stream>>>(slot_token, p_accum);

    transpose_convert_kernel<<<dim3(FF_ / 64, C_ / 64, E_), 256, 0, stream>>>(W1, W1T, C_, FF_);
    transpose_convert_kernel<<<dim3(C_ / 64, FF_ / 64, E_), 256, 0, stream>>>(W2, W2T, FF_, C_);

    router_kernel<<<BT_ / 8, 256, 0, stream>>>(x, Wc, bc, gates, topk, sc0, p_accum);
    sort_kernel<<<1, 1024, 0, stream>>>(sc0, order);
    assign_kernel<<<1, 1024, 0, stream>>>(order, topk, token_slot, slot_token, counts);
    gather_kernel<<<E_ * CAP_, 256, 0, stream>>>(x, slot_token, expert_in);

    gemm_bf16_kernel<0><<<dim3(FF_ / 128, CAP_ / 128, E_), 256, 0, stream>>>(
        expert_in, W1T, b1, (void*)hbuf, FF_, C_,
        (long)CAP_ * C_, (long)FF_ * C_, (long)CAP_ * FF_, FF_);
    gemm_bf16_kernel<1><<<dim3(C_ / 128, CAP_ / 128, E_), 256, 0, stream>>>(
        hbuf, W2T, b2, (void*)eout, C_, FF_,
        (long)CAP_ * FF_, (long)C_ * FF_, (long)CAP_ * C_, C_);

    combine_kernel<<<BT_, 256, 0, stream>>>(eout, token_slot, gates, out);
    stats_kernel<<<1, 32, 0, stream>>>(counts, p_accum, tail);
}
